// MaskMatteDynAttenHead_1949915152911
// MI455X (gfx1250) — matrix operand order repaired and checked by reading, not compiled, not run on hardware
//
#include <hip/hip_runtime.h>
#include <math.h>

// ============================================================================
// MI455X (gfx1250) implementation.
// All heavy matmuls (feat_proj, MHA projections, attention, conv1 as 9 shifted
// GEMMs, conv2, final einsum) go through v_wmma_f32_16x16x32_f16 (f16 in, f32
// accum). Small reductions (layernorm/softmax/pool/topk) are wave32 VALU.
// ============================================================================

typedef __attribute__((ext_vector_type(16))) _Float16 v16h;
typedef __attribute__((ext_vector_type(2)))  __fp16   fp16x2;
typedef __attribute__((ext_vector_type(8)))  float    v8f;
union HFrag { uint4 u[2]; v16h v; };
union PackCvt { fp16x2 h; unsigned u; };

#define LPU 20  // LDS pitch in dwords (= 40 halves = 80B: 16B aligned, staggered)

struct GemmP {
  const float* A; const float* B; float* D; const float* bias;
  long long aRow, aCol, bRow, bCol, dRow, dCol;
  int M, N, K, bz1, flags;
  long long aB1, aB2, bB1, bB2, dB1, dB2;
};
enum { GF_RELU = 1, GF_ACC = 2 };

// ---------------------------------------------------------------------------
// Generic batched GEMM: D[m,n] = sum_k A[m,k]*B[n,k] (+bias[n]) (+D) (relu)
// Block tile 64(M) x 128(N), 8 waves in 2x4; each wave: 2x2 WMMA tiles.
// K-step 32 staged via LDS f16 (branch-free clamped loads, packed cvt).
// ---------------------------------------------------------------------------
__global__ __launch_bounds__(256) void k_gemm(GemmP p) {
  const int z  = blockIdx.z;
  const int z1 = p.bz1 > 1 ? (z % p.bz1) : 0;
  const int z2 = p.bz1 > 1 ? (z / p.bz1) : z;
  const float* A = p.A + z1 * p.aB1 + z2 * p.aB2;
  const float* B = p.B + z1 * p.bB1 + z2 * p.bB2;
  float*       D = p.D + z1 * p.dB1 + z2 * p.dB2;
  const int m0 = blockIdx.y * 64, n0 = blockIdx.x * 128;

  __shared__ __align__(16) unsigned As32[64 * LPU];    // 64 rows x 32 halves
  __shared__ __align__(16) unsigned Bs32[128 * LPU];   // 128 rows x 32 halves

  const int t = threadIdx.x, lane = t & 31, wid = t >> 5;
  const int half = lane >> 4, lm = lane & 15;
  const int wm = wid & 1, wn = wid >> 1;               // 2 m-waves x 4 n-waves

  // ---- precomputed staging geometry (branch-free, always-in-bounds) ----
  const int kp = (t & 15) * 2;        // this thread's fixed k-pair within step
  const int rb = t >> 4;              // base row 0..15
  long long aRO[4]; bool aOK[4];
  #pragma unroll
  for (int j = 0; j < 4; ++j) {
    int r = rb + 16 * j;
    aOK[j] = (m0 + r) < p.M;
    int rc = m0 + r; if (rc > p.M - 1) rc = p.M - 1;
    aRO[j] = (long long)rc * p.aRow;
  }
  long long bRO[8]; bool bOK[8];
  #pragma unroll
  for (int j = 0; j < 8; ++j) {
    int r = rb + 16 * j;
    bOK[j] = (n0 + r) < p.N;
    int rc = n0 + r; if (rc > p.N - 1) rc = p.N - 1;
    bRO[j] = (long long)rc * p.bRow;
  }

  v8f acc[2][2];
  acc[0][0] = v8f{}; acc[0][1] = v8f{}; acc[1][0] = v8f{}; acc[1][1] = v8f{};

  for (int k0 = 0; k0 < p.K; k0 += 32) {
    // ---- stage A/B tiles (f32 -> packed f16), zero-fill out of range ----
    int kc0 = k0 + kp;     if (kc0 > p.K - 1) kc0 = p.K - 1;
    int kc1 = k0 + kp + 1; if (kc1 > p.K - 1) kc1 = p.K - 1;
    const bool kv0 = (k0 + kp) < p.K, kv1 = (k0 + kp + 1) < p.K;
    const long long aK0 = (long long)kc0 * p.aCol, aK1 = (long long)kc1 * p.aCol;
    const long long bK0 = (long long)kc0 * p.bCol, bK1 = (long long)kc1 * p.bCol;

    float av0[4], av1[4], bv0[8], bv1[8];
    #pragma unroll
    for (int j = 0; j < 4; ++j) { av0[j] = A[aRO[j] + aK0]; av1[j] = A[aRO[j] + aK1]; }
    #pragma unroll
    for (int j = 0; j < 8; ++j) { bv0[j] = B[bRO[j] + bK0]; bv1[j] = B[bRO[j] + bK1]; }
    #pragma unroll
    for (int j = 0; j < 4; ++j) {
      PackCvt c;
      c.h = __builtin_amdgcn_cvt_pkrtz((aOK[j] && kv0) ? av0[j] : 0.f,
                                       (aOK[j] && kv1) ? av1[j] : 0.f);
      As32[(rb + 16 * j) * LPU + (t & 15)] = c.u;
    }
    #pragma unroll
    for (int j = 0; j < 8; ++j) {
      PackCvt c;
      c.h = __builtin_amdgcn_cvt_pkrtz((bOK[j] && kv0) ? bv0[j] : 0.f,
                                       (bOK[j] && kv1) ? bv1[j] : 0.f);
      Bs32[(rb + 16 * j) * LPU + (t & 15)] = c.u;
    }
    __syncthreads();

    // ---- fragments (ISA 7.12.2 16-bit layouts) ----
    // A: lane<16 rows m, K = {0..7,16..23}; lane>=16 K = {8..15,24..31}
    HFrag a[2];
    #pragma unroll
    for (int i = 0; i < 2; ++i) {
      int ar = wm * 32 + i * 16 + lm;
      a[i].u[0] = *(const uint4*)&As32[ar * LPU + (half << 2)];
      a[i].u[1] = *(const uint4*)&As32[ar * LPU + 8 + (half << 2)];
    }
    // B: col = lane%16, lane<16 K = {0..7,16..23}; lane>=16 K = {8..15,24..31}
    HFrag b[2];
    #pragma unroll
    for (int j = 0; j < 2; ++j) {
      int br = wn * 32 + j * 16 + lm;
      b[j].u[0] = *(const uint4*)&Bs32[br * LPU + (half << 2)];
      b[j].u[1] = *(const uint4*)&Bs32[br * LPU + 8 + (half << 2)];
    }
    #pragma unroll
    for (int i = 0; i < 2; ++i)
      #pragma unroll
      for (int j = 0; j < 2; ++j)
        acc[i][j] = __builtin_amdgcn_wmma_f32_16x16x32_f16(
            false, a[i].v, false, b[j].v, (short)0, acc[i][j], false, false);
    __syncthreads();
  }

  // ---- store: C/D layout: vgpr r -> m = r + 8*half, n = lane%16 ----
  #pragma unroll
  for (int i = 0; i < 2; ++i) {
    #pragma unroll
    for (int r = 0; r < 8; ++r) {
      int gm = m0 + wm * 32 + i * 16 + half * 8 + r;
      if (gm >= p.M) continue;
      #pragma unroll
      for (int j = 0; j < 2; ++j) {
        int gn = n0 + wn * 32 + j * 16 + lm;
        if (gn >= p.N) continue;
        float v = acc[i][j][r];
        if (p.bias) v += p.bias[gn];
        long long o = (long long)gm * p.dRow + (long long)gn * p.dCol;
        if (p.flags & GF_ACC) v += D[o];
        if (p.flags & GF_RELU) v = fmaxf(v, 0.f);
        D[o] = v;
      }
    }
  }
}

// ---------------------------------------------------------------------------
// Positional encoding pe3d value at (c, f, y, x); C=256, npf=128, t=4, H=W=128
// ---------------------------------------------------------------------------
__device__ __forceinline__ float pe_val(int c, int f, int y, int x) {
  const float TWO_PI = 6.28318530717958647692f;
  float zs = (float)(f + 1) / (4.0f + 1e-6f) * TWO_PI;
  float dz = __powf(10000.f, 2.f * (float)(c >> 1) / 256.f);
  float az = zs / dz;
  float pz = (c & 1) ? __cosf(az) : __sinf(az);
  int  cc = (c < 128) ? c : c - 128;
  float v = ((c < 128) ? (float)(y + 1) : (float)(x + 1)) / (128.f + 1e-6f) * TWO_PI;
  float d = __powf(10000.f, 2.f * (float)(cc >> 1) / 128.f);
  float a = v / d;
  float pp = (cc & 1) ? __cosf(a) : __sinf(a);
  return pp + pz;
}

// 16x16 avg pool: qf[img][c][cell], qpp = pool(pe*mask)/(mr+1e-6), mr[img][cell]
__global__ __launch_bounds__(256) void k_pool(const float* feat, const float* mask,
                                              float* qf, float* qpp, float* mr) {
  int cell = blockIdx.x, img = blockIdx.y;
  int cy = cell >> 3, cx = cell & 7, f = img & 3;
  int py0 = cy * 16, px0 = cx * 16;
  __shared__ float mwin[256];
  __shared__ float mrs;
  int t = threadIdx.x;
  mwin[t] = mask[(size_t)img * 16384 + (py0 + (t >> 4)) * 128 + px0 + (t & 15)];
  __syncthreads();
  if (t == 0) {
    float s = 0.f;
    for (int i = 0; i < 256; ++i) s += mwin[i];
    mrs = s * (1.f / 256.f);
    mr[img * 64 + cell] = mrs;
  }
  __syncthreads();
  int c = t;
  float sf = 0.f, sp = 0.f;
  const float* fp = feat + ((size_t)img * 256 + c) * 16384;
  for (int i = 0; i < 256; ++i) {
    int y = py0 + (i >> 4), x = px0 + (i & 15);
    sf += fp[y * 128 + x];
    sp += pe_val(c, f, y, x) * mwin[i];
  }
  qf[((size_t)img * 256 + c) * 64 + cell]  = sf * (1.f / 256.f);
  qpp[((size_t)img * 256 + c) * 64 + cell] = (sp * (1.f / 256.f)) / (mrs + 1e-6f);
}

// fpos[s][fb][c] = pe(c, f, y, x), fb = f*2+bb (independent of bb)
__global__ __launch_bounds__(256) void k_fpos(float* fpos, size_t n) {
  size_t o = (size_t)blockIdx.x * 256 + threadIdx.x;
  if (o >= n) return;
  int c = o & 255; int fb = (o >> 8) & 7; int s = (int)(o >> 11);
  fpos[o] = pe_val(c, fb >> 1, s >> 7, s & 127);
}

// top-8 of 64 per image (descending, first-index ties like jax top_k)
__global__ void k_topk(const float* mr, float* tkv, int* tki) {
  int img = threadIdx.x;
  if (img >= 8) return;
  float v[64];
  for (int i = 0; i < 64; ++i) v[i] = mr[img * 64 + i];
  for (int j = 0; j < 8; ++j) {
    float best = -1e30f; int bi = 0;
    for (int i = 0; i < 64; ++i) if (v[i] > best) { best = v[i]; bi = i; }
    tkv[img * 8 + j] = best; tki[img * 8 + j] = bi; v[bi] = -1e30f;
  }
}

__global__ __launch_bounds__(256) void k_gather_tok(const float* qf, const float* qpp,
    const int* tki, float* tokG, float* tposG) {
  int o = blockIdx.x * 256 + threadIdx.x;
  if (o >= 16384) return;
  int c = o & 255, row = o >> 8, img = row >> 3;
  int cell = tki[row];
  tokG[o]  = qf[((size_t)img * 256 + c) * 64 + cell];
  tposG[o] = qpp[((size_t)img * 256 + c) * 64 + cell];
}

// tok -> sequence layout row (q*4+f)*2+bb, add mask-ratio embedding
__global__ __launch_bounds__(256) void k_tok_finalize(const float* tokP, const float* tposG,
    const float* tkv, const float* mrw, const float* mrb, float* tokS, float* tposS) {
  int o = blockIdx.x * 256 + threadIdx.x;
  if (o >= 16384) return;
  int c = o & 255, row = o >> 8, img = row >> 3, q = row & 7;
  int dst = (q * 4 + (img & 3)) * 2 + (img >> 2);
  tokS[dst * 256 + c]  = tokP[o] + tkv[row] * mrw[c] + mrb[c];
  tposS[dst * 256 + c] = tposG[o];
}

// ori_feat [img][c][s] -> featv_raw [s][fb][c]   (fb = f*2+bb, img = bb*4+f)
__global__ __launch_bounds__(256) void k_pack_featv(const float* ori, float* dst, size_t n) {
  size_t o = (size_t)blockIdx.x * 256 + threadIdx.x;
  if (o >= n) return;
  int c = o & 255; int fb = (o >> 8) & 7; int s = (int)(o >> 11);
  int img = (fb & 1) * 4 + (fb >> 1);
  dst[o] = ori[((size_t)img * 256 + c) * 16384 + s];
}

// featv [s][fb][c] -> channels-last images xcl [img][s][c]
__global__ __launch_bounds__(256) void k_pack_xcl(const float* featv, float* xcl, size_t n) {
  size_t o = (size_t)blockIdx.x * 256 + threadIdx.x;
  if (o >= n) return;
  int c = o & 255; int s = (int)((o >> 8) & 16383); int img = (int)(o >> 22);
  int fb = (img & 3) * 2 + (img >> 2);
  xcl[o] = featv[((size_t)(s * 8 + fb)) * 256 + c];
}

// shifted copy with zero pad for one 3x3 tap
__global__ __launch_bounds__(256) void k_shift(const float* x, float* y, int dy, int dx, size_t n) {
  size_t o = (size_t)blockIdx.x * 256 + threadIdx.x;
  if (o >= n) return;
  int c = o & 255; int s = (int)((o >> 8) & 16383); int img = (int)(o >> 22);
  int yy = (s >> 7) + dy - 1, xx = (s & 127) + dx - 1;
  float v = 0.f;
  if (yy >= 0 && yy < 128 && xx >= 0 && xx < 128)
    v = x[((size_t)img << 22) + ((size_t)(yy * 128 + xx) << 8) + c];
  y[o] = v;
}

__global__ __launch_bounds__(256) void k_add(const float* a, const float* b, float* o, size_t n) {
  size_t i = (size_t)blockIdx.x * 256 + threadIdx.x;
  if (i < n) o[i] = a[i] + b[i];
}

__global__ __launch_bounds__(256) void k_bnleaky(float* x, const float* g, const float* b,
    const float* mean, const float* var, size_t n) {
  size_t i = (size_t)blockIdx.x * 256 + threadIdx.x;
  if (i >= n) return;
  int c = i & 255;
  float v = (x[i] - mean[c]) * rsqrtf(var[c] + 1e-5f) * g[c] + b[c];
  x[i] = (v >= 0.f) ? v : 0.2f * v;
}

// wave-per-row layernorm over C=256; optional residual; may run in place
__global__ __launch_bounds__(256) void k_layernorm(const float* x, const float* resid,
    const float* g, const float* b, float* out, int rows) {
  int row = blockIdx.x * 8 + (threadIdx.x >> 5);
  if (row >= rows) return;
  int lane = threadIdx.x & 31;
  const float* xr = x + (long long)row * 256;
  const float* rr = resid ? resid + (long long)row * 256 : nullptr;
  float v[8]; float s = 0.f, sq = 0.f;
  #pragma unroll
  for (int j = 0; j < 8; ++j) {
    int c = lane + j * 32;
    float t = xr[c] + (rr ? rr[c] : 0.f);
    v[j] = t; s += t; sq += t * t;
  }
  #pragma unroll
  for (int off = 16; off; off >>= 1) { s += __shfl_xor(s, off, 32); sq += __shfl_xor(sq, off, 32); }
  float m = s * (1.f / 256.f);
  float rstd = rsqrtf(sq * (1.f / 256.f) - m * m + 1e-5f);
  float* orow = out + (long long)row * 256;
  #pragma unroll
  for (int j = 0; j < 8; ++j) {
    int c = lane + j * 32;
    orow[c] = (v[j] - m) * rstd * g[c] + b[c];
  }
}

// block-per-row softmax (long rows, e.g. 16384)
__global__ __launch_bounds__(256) void k_softmax_big(float* base, long long batchStride,
    long long rowStride, int cols, float scale) {
  float* p = base + blockIdx.y * batchStride + blockIdx.x * rowStride;
  __shared__ float red[256];
  int t = threadIdx.x;
  float m = -1e30f;
  for (int c = t; c < cols; c += 256) m = fmaxf(m, p[c] * scale);
  red[t] = m; __syncthreads();
  for (int o = 128; o; o >>= 1) { if (t < o) red[t] = fmaxf(red[t], red[t + o]); __syncthreads(); }
  m = red[0]; __syncthreads();
  float s = 0.f;
  for (int c = t; c < cols; c += 256) { float e = __expf(p[c] * scale - m); p[c] = e; s += e; }
  red[t] = s; __syncthreads();
  for (int o = 128; o; o >>= 1) { if (t < o) red[t] += red[t + o]; __syncthreads(); }
  float inv = 1.f / red[0];
  for (int c = t; c < cols; c += 256) p[c] *= inv;
}

// thread-per-row softmax (short rows <= 32)
__global__ __launch_bounds__(256) void k_softmax_tiny(float* base, long long rows,
    int cols, int ld, float scale) {
  long long r = (long long)blockIdx.x * 256 + threadIdx.x;
  if (r >= rows) return;
  float* p = base + r * ld;
  float m = -1e30f;
  for (int c = 0; c < cols; ++c) m = fmaxf(m, p[c] * scale);
  float s = 0.f;
  for (int c = 0; c < cols; ++c) s += __expf(p[c] * scale - m);
  float inv = 1.f / s;
  for (int c = 0; c < cols; ++c) p[c] = __expf(p[c] * scale - m) * inv;
}

// sequence-layout tokens -> image-major (img*8+q) rows
__global__ __launch_bounds__(256) void k_reorder_tokf(const float* src, float* dst) {
  int o = blockIdx.x * 256 + threadIdx.x;
  if (o >= 16384) return;
  int c = o & 255, r = o >> 8, img = r >> 3, q = r & 7;
  int s = (q * 4 + (img & 3)) * 2 + (img >> 2);
  dst[o] = src[s * 256 + c];
}

// 3x3 query conv over 8 query channels -> 1 output channel
__global__ __launch_bounds__(256) void k_qconv(const float* om, const float* qw,
    const float* qb, float* out) {
  int o = blockIdx.x * 256 + threadIdx.x;
  if (o >= 131072) return;
  int img = o >> 14, s = o & 16383, y = s >> 7, x = s & 127;
  float acc = qb[0];
  for (int dy = 0; dy < 3; ++dy) {
    int yy = y + dy - 1; if (yy < 0 || yy >= 128) continue;
    for (int dx = 0; dx < 3; ++dx) {
      int xx = x + dx - 1; if (xx < 0 || xx >= 128) continue;
      const float* p = om + ((size_t)img << 17) + ((size_t)(yy * 128 + xx) << 3);
      #pragma unroll
      for (int q = 0; q < 8; ++q) acc += p[q] * qw[(q * 3 + dy) * 3 + dx];
    }
  }
  out[o] = acc;
}

// ============================ host side ====================================
static inline int cdiv(long long a, long long b) { return (int)((a + b - 1) / b); }

static inline void gemm(hipStream_t st,
    const float* A, long long aRow, long long aCol,
    const float* B, long long bRow, long long bCol,
    float* D, long long dRow, long long dCol,
    const float* bias, int M, int N, int K, int flags = 0,
    int nz = 1, int bz1 = 1,
    long long aB1 = 0, long long aB2 = 0, long long bB1 = 0, long long bB2 = 0,
    long long dB1 = 0, long long dB2 = 0) {
  GemmP p;
  p.A = A; p.B = B; p.D = D; p.bias = bias;
  p.aRow = aRow; p.aCol = aCol; p.bRow = bRow; p.bCol = bCol; p.dRow = dRow; p.dCol = dCol;
  p.M = M; p.N = N; p.K = K; p.bz1 = bz1 < 1 ? 1 : bz1; p.flags = flags;
  p.aB1 = aB1; p.aB2 = aB2; p.bB1 = bB1; p.bB2 = bB2; p.dB1 = dB1; p.dB2 = dB2;
  dim3 grid(cdiv(N, 128), cdiv(M, 64), nz);
  k_gemm<<<grid, 256, 0, st>>>(p);
}

extern "C" void kernel_launch(void* const* d_in, const int* in_sizes, int n_in,
                              void* d_out, int out_size, void* d_ws, size_t ws_size,
                              hipStream_t stream) {
  auto IN = [&](int i) { return (const float*)d_in[i]; };
  const float* ori  = IN(0);
  const float* mask = IN(1);
  // params flattened in dict order; blocks: stride 24 starting at index 8
  auto BI = [&](int blk, int off) { return IN(8 + blk * 24 + off); };

  const size_t BIG = 131072ull * 256ull;       // 33,554,432 elems
  float* W = (float*)d_ws;
  float* featv = W;                 // current featv (updated by ca2)
  float* fpos  = W + BIG;
  float* fvp   = W + 2 * BIG;       // featv + fpos (also xcl in conv stage)
  float* bufA  = W + 3 * BIG;       // kp / qp2 / shifted conv input
  float* bufB  = W + 4 * BIG;       // vp / attn-out concat / conv accumulator
  float* SC    = W + 5 * BIG;       // scores: 32 * 16 * 16384 = 8,388,608
  float* sp    = SC + 32ull * 16 * 16384;
  auto alloc = [&](size_t n) { float* r = sp; sp += n; return r; };
  float* qf    = alloc(131072);
  float* qpp   = alloc(131072);
  float* mr    = alloc(512);
  float* tkv   = alloc(64);
  float* tokG  = alloc(16384);
  float* tposG = alloc(16384);
  float* tokP  = alloc(16384);
  float* tokS  = alloc(16384);
  float* tposS = alloc(16384);
  float* qkbuf = alloc(16384);
  float* p1    = alloc(16384);
  float* p2    = alloc(16384);
  float* p3    = alloc(16384);
  float* sas   = alloc(8192);
  float* s1    = alloc(16384);
  float* s2    = alloc(16384);
  float* tokfp = alloc(16384);
  float* tokfr = alloc(16384);
  float* tokf  = alloc(16384);
  float* omcl  = alloc(8ull * 16384 * 8);
  int*   tki   = (int*)alloc(64);

  const long long SB = 16ll * 16384;           // SC per-batch stride (h step)

  // ---------------- stage 0: pools, PE, tokens, featv projection -----------
  k_pool<<<dim3(64, 8), 256, 0, stream>>>(ori, mask, qf, qpp, mr);
  k_fpos<<<cdiv(BIG, 256), 256, 0, stream>>>(fpos, BIG);
  k_topk<<<1, 8, 0, stream>>>(mr, tkv, tki);
  k_gather_tok<<<64, 256, 0, stream>>>(qf, qpp, tki, tokG, tposG);
  gemm(stream, tokG, 256, 1, IN(4), 256, 1, tokP, 256, 1, IN(5), 64, 256, 256);
  k_tok_finalize<<<64, 256, 0, stream>>>(tokP, tposG, tkv, IN(6), IN(7), tokS, tposS);
  k_pack_featv<<<cdiv(BIG, 256), 256, 0, stream>>>(ori, fvp, BIG);
  gemm(stream, fvp, 256, 1, IN(2), 256, 1, featv, 256, 1, IN(3), 131072, 256, 256);

  // cross-attention tok->featv (used by ca1 of each block and final_ca)
  auto cross_t2f = [&](const float* in_w, const float* in_b,
                       const float* out_w, const float* out_b,
                       const float* g, const float* b) {
    k_add<<<64, 256, 0, stream>>>(tokS, tposS, qkbuf, 16384);
    gemm(stream, qkbuf, 256, 1, in_w, 256, 1, p1, 256, 1, in_b, 64, 256, 256);
    gemm(stream, fvp, 256, 1, in_w + 65536, 256, 1, bufA, 256, 1, in_b + 256, 131072, 256, 256);
    gemm(stream, featv, 256, 1, in_w + 131072, 256, 1, bufB, 256, 1, in_b + 512, 131072, 256, 256);
    // scores[fb][h][l][s], z = fb*4 + h
    gemm(stream, p1, 2048, 1, bufA, 2048, 1, SC, 16384, 1, nullptr,
         8, 16384, 64, 0, 32, 4, 64, 256, 64, 256, SB, 4 * SB);
    k_softmax_big<<<dim3(8, 32), 256, 0, stream>>>(SC, SB, 16384, 16384, 0.125f);
    gemm(stream, SC, 16384, 1, bufB, 1, 2048, s1, 2048, 1, nullptr,
         8, 64, 16384, 0, 32, 4, SB, 4 * SB, 64, 256, 64, 256);
    gemm(stream, s1, 256, 1, out_w, 256, 1, s2, 256, 1, out_b, 64, 256, 256);
    k_layernorm<<<8, 256, 0, stream>>>(s2, tokS, g, b, tokS, 64);
  };

  for (int blk = 0; blk < 2; ++blk) {
    const float* sa_w  = BI(blk, 0),  *sa_b  = BI(blk, 1);
    const float* sa_ow = BI(blk, 2),  *sa_ob = BI(blk, 3);
    const float* c2_w  = BI(blk, 18), *c2_b  = BI(blk, 19);
    const float* c2_ow = BI(blk, 20), *c2_ob = BI(blk, 21);

    k_add<<<cdiv(BIG, 256), 256, 0, stream>>>(featv, fpos, fvp, BIG);  // featv+pos

    // ---- self attention on tokens (L=32, B=2) ----
    k_add<<<64, 256, 0, stream>>>(tokS, tposS, qkbuf, 16384);
    gemm(stream, qkbuf, 256, 1, sa_w, 256, 1, p1, 256, 1, sa_b, 64, 256, 256);
    gemm(stream, qkbuf, 256, 1, sa_w + 65536, 256, 1, p2, 256, 1, sa_b + 256, 64, 256, 256);
    gemm(stream, tokS, 256, 1, sa_w + 131072, 256, 1, p3, 256, 1, sa_b + 512, 64, 256, 256);
    gemm(stream, p1, 512, 1, p2, 512, 1, sas, 32, 1, nullptr,
         32, 32, 64, 0, 8, 4, 64, 256, 64, 256, 1024, 4096);   // z = bb*4+h
    k_softmax_tiny<<<1, 256, 0, stream>>>(sas, 256, 32, 32, 0.125f);
    gemm(stream, sas, 32, 1, p3, 1, 512, s1, 512, 1, nullptr,
         32, 64, 32, 0, 8, 4, 1024, 4096, 64, 256, 64, 256);
    gemm(stream, s1, 256, 1, sa_ow, 256, 1, s2, 256, 1, sa_ob, 64, 256, 256);
    k_layernorm<<<8, 256, 0, stream>>>(s2, tokS, BI(blk, 4), BI(blk, 5), tokS, 64);

    // ---- ca1: tokens attend to featv ----
    cross_t2f(BI(blk, 6), BI(blk, 7), BI(blk, 8), BI(blk, 9), BI(blk, 10), BI(blk, 11));

    // ---- ffn ----
    gemm(stream, tokS, 256, 1, BI(blk, 12), 256, 1, p1, 256, 1, BI(blk, 13), 64, 256, 256, GF_RELU);
    gemm(stream, p1, 256, 1, BI(blk, 14), 256, 1, s2, 256, 1, BI(blk, 15), 64, 256, 256);
    k_layernorm<<<8, 256, 0, stream>>>(s2, tokS, BI(blk, 16), BI(blk, 17), tokS, 64);

    // ---- ca2: featv attends to tokens ----
    gemm(stream, fvp, 256, 1, c2_w, 256, 1, bufA, 256, 1, c2_b, 131072, 256, 256); // qp2
    k_add<<<64, 256, 0, stream>>>(tokS, tposS, qkbuf, 16384);
    gemm(stream, qkbuf, 256, 1, c2_w + 65536, 256, 1, p2, 256, 1, c2_b + 256, 64, 256, 256);
    gemm(stream, tokS, 256, 1, c2_w + 131072, 256, 1, p3, 256, 1, c2_b + 512, 64, 256, 256);
    gemm(stream, bufA, 2048, 1, p2, 2048, 1, SC, 16, 1, nullptr,
         16384, 8, 64, 0, 32, 4, 64, 256, 64, 256, SB, 4 * SB);
    k_softmax_tiny<<<cdiv(32ll * 16384, 256), 256, 0, stream>>>(SC, 32ll * 16384, 8, 16, 0.125f);
    gemm(stream, SC, 16, 1, p3, 1, 2048, bufB, 2048, 1, nullptr,
         16384, 64, 8, 0, 32, 4, SB, 4 * SB, 64, 256, 64, 256);
    gemm(stream, bufB, 256, 1, c2_ow, 256, 1, fvp, 256, 1, c2_ob, 131072, 256, 256);
    k_layernorm<<<16384, 256, 0, stream>>>(fvp, featv, BI(blk, 22), BI(blk, 23), featv, 131072);
  }

  // ---------------- final cross attention ----------------------------------
  k_add<<<cdiv(BIG, 256), 256, 0, stream>>>(featv, fpos, fvp, BIG);
  cross_t2f(IN(56), IN(57), IN(58), IN(59), IN(60), IN(61));

  // ---------------- decoder head -------------------------------------------
  gemm(stream, tokS, 256, 1, IN(62), 256, 1, tokfp, 256, 1, IN(63), 64, 256, 256);
  k_reorder_tokf<<<64, 256, 0, stream>>>(tokfp, tokfr);
  k_layernorm<<<8, 256, 0, stream>>>(tokfr, nullptr, IN(64), IN(65), tokf, 64);

  // conv1 (3x3 SAME) as 9 shifted accumulating WMMA GEMMs, channels-last
  k_pack_xcl<<<cdiv(BIG, 256), 256, 0, stream>>>(featv, fvp, BIG);
  for (int tap = 0; tap < 9; ++tap) {
    k_shift<<<cdiv(BIG, 256), 256, 0, stream>>>(fvp, bufA, tap / 3, tap % 3, BIG);
    gemm(stream, bufA, 256, 1, IN(66) + tap, 2304, 9, bufB, 256, 1, nullptr,
         131072, 256, 256, tap ? GF_ACC : 0);
  }
  k_bnleaky<<<cdiv(BIG, 256), 256, 0, stream>>>(bufB, IN(67), IN(68), IN(69), IN(70), BIG);
  // conv2 (1x1) + bn2 + leaky
  gemm(stream, bufB, 256, 1, IN(71), 256, 1, fvp, 256, 1, nullptr, 131072, 256, 256);
  k_bnleaky<<<cdiv(BIG, 256), 256, 0, stream>>>(fvp, IN(72), IN(73), IN(74), IN(75), BIG);
  // om[img][pix][q] = x_cl . tokf^T  (per-image GEMM, N=8)
  gemm(stream, fvp, 256, 1, tokf, 256, 1, omcl, 8, 1, nullptr,
       16384, 8, 256, 0, 8, 8, 16384ll * 256, 0, 2048, 0, 131072, 0);
  // 3x3 query conv -> final output (8,1,128,128)
  k_qconv<<<512, 256, 0, stream>>>(omcl, IN(76), IN(77), (float*)d_out);
}
